// Attention_46901042872339
// MI455X (gfx1250) — matrix operand order repaired and checked by reading, not compiled, not run on hardware
//
#include <hip/hip_runtime.h>

typedef __attribute__((ext_vector_type(16))) _Float16 v16h;
typedef __attribute__((ext_vector_type(8)))  float    v8f;
typedef int v4i __attribute__((vector_size(16)));   // matches async-LDS builtin param

#define NN 256   // nodes
#define VD 40    // value feature dim
#define QKD 50   // q/k dim
#define VP 64    // padded feature / qk dim

union TileU { v16h h; unsigned int w[8]; };

// A-operand (16x32 f16) tile from row-major f16 LDS.
// ISA layout: lanes 0-15 & 16-31 both hold M=lane%16; VGPR v holds K pair:
//   k = (v<4 ? 2v : 16+2(v-4)) + (lane>=16 ? 8 : 0)
static __device__ inline v16h ld_tile_a(const _Float16* base, int row0, int stride,
                                        int kb, int lane) {
  int r  = row0 + (lane & 15);
  int ko = kb + ((lane & 16) ? 8 : 0);
  TileU u;
#pragma unroll
  for (int v = 0; v < 8; ++v) {
    int k = ko + ((v < 4) ? (2 * v) : (16 + 2 * (v - 4)));
    u.w[v] = *(const unsigned int*)(base + r * stride + k);
  }
  return u.h;
}

// B-operand (32x16 f16) tile: lane%16 selects column n; we read B[k][n] as
// row n of a row-major f16 LDS array. Same K layout as the A tile:
//   k = (v<4 ? 2v : 16+2(v-4)) + (lane>=16 ? 8 : 0)
static __device__ inline v16h ld_tile_b(const _Float16* base, int col0, int stride,
                                        int kb, int lane) {
  int r  = col0 + (lane & 15);
  int ko = kb + ((lane & 16) ? 8 : 0);
  TileU u;
#pragma unroll
  for (int v = 0; v < 8; ++v) {
    u.w[v] = *(const unsigned int*)(base + r * stride + ko + ((v < 4) ? (2 * v) : (16 + 2 * (v - 4))));
  }
  return u.h;
}

static __device__ inline v8f wmma16(v16h a, v16h b, v8f c) {
  return __builtin_amdgcn_wmma_f32_16x16x32_f16(false, a, false, b, (short)0, c,
                                                false, false);
}

// CDNA5 has a hardware TANH transcendental; avoid the OCML polynomial.
static __device__ inline float fast_tanh(float x) {
#if __has_builtin(__builtin_amdgcn_tanhf)
  return __builtin_amdgcn_tanhf(x);
#elif __has_builtin(__builtin_amdgcn_tanh_f32)
  return __builtin_amdgcn_tanh_f32(x);
#else
  return tanhf(x);
#endif
}

__global__ __launch_bounds__(512, 1)
void gat_mp_kernel(const float* __restrict__ values,
                   const float* __restrict__ adj,
                   const float* __restrict__ Wq,
                   const float* __restrict__ bqg,
                   const float* __restrict__ Wk,
                   const float* __restrict__ bkg,
                   float* __restrict__ out) {
  extern __shared__ char smem[];
  _Float16* valh = (_Float16*)(smem);            // 256x64 row-major   (32KB)
  _Float16* valt = (_Float16*)(smem + 32768);    // 64x256 transposed  (32KB)
  _Float16* qh   = (_Float16*)(smem + 65536);    // 256x64             (32KB)
  _Float16* kh   = (_Float16*)(smem + 98304);    // 256x64             (32KB)
  _Float16* wqh  = (_Float16*)(smem + 131072);   // 64x64 [qk][v]      (8KB)
  _Float16* wkh  = (_Float16*)(smem + 139264);   // 64x64              (8KB)
  _Float16* alph = (_Float16*)(smem + 147456);   // 256x256            (128KB)
  unsigned int* maskw = (unsigned int*)(smem + 278528); // 256x8 bits  (8KB)
  float* bqs = (float*)(smem + 286720);          // 64
  float* bks = (float*)(smem + 286976);          // 64  (total 287232 B)

  const int tid  = threadIdx.x;
  const int lane = tid & 31;
  const int wid  = tid >> 5;          // 16 waves
  const int f    = blockIdx.x;
  const int slab = wid * 16;          // 16 rows per wave
  const int nloc = lane & 15;
  const int hsel = lane >> 4;

  const float* vin  = values + (size_t)f * (NN * VD);
  const float* adjf = adj + (size_t)f * (NN * NN);

  // ---------------- setup ----------------
  // Stage this graph's values block (40KB f32) into the idle alpha region
  // via async global->LDS copies, overlapped with the zero/fill work below.
  float* stagef = (float*)alph;
#if __has_builtin(__builtin_amdgcn_global_load_async_to_lds_b128)
  for (int c = tid; c < (NN * VD) / 4; c += 512) {
    __builtin_amdgcn_global_load_async_to_lds_b128(
        (__attribute__((address_space(1))) v4i*)(vin + c * 4),
        (__attribute__((address_space(3))) v4i*)(stagef + c * 4), 0, 0);
  }
#else
  for (int c = tid; c < (NN * VD) / 4; c += 512) {
    *(float4*)(stagef + c * 4) = *(const float4*)(vin + c * 4);
  }
#endif

  for (int i = tid; i < NN * VP; i += 512) valh[i] = (_Float16)0.f;
  for (int i = tid; i < VP * NN; i += 512) valt[i] = (_Float16)0.f;
  for (int i = tid; i < VP * VP; i += 512) { wqh[i] = (_Float16)0.f; wkh[i] = (_Float16)0.f; }
  if (tid < 64) {
    bqs[tid] = (tid < QKD) ? bqg[tid] : 0.f;
    bks[tid] = (tid < QKD) ? bkg[tid] : 0.f;
  }

#if __has_builtin(__builtin_amdgcn_global_load_async_to_lds_b128)
#if __has_builtin(__builtin_amdgcn_s_wait_asynccnt)
  __builtin_amdgcn_s_wait_asynccnt(0);
#else
  asm volatile("s_wait_asynccnt 0" ::: "memory");
#endif
#endif
  __syncthreads();

  // fill values (f16 row-major + transposed) from staged copy, and weights
  for (int i = tid; i < NN * VD; i += 512) {
    int n = i / VD, v = i - n * VD;
    float x = stagef[i];
    valh[n * VP + v] = (_Float16)x;
    valt[v * NN + n] = (_Float16)x;
  }
  for (int i = tid; i < QKD * VD; i += 512) {
    int q = i / VD, v = i - q * VD;
    wqh[q * VP + v] = (_Float16)Wq[i];
    wkh[q * VP + v] = (_Float16)Wk[i];
  }
  // adjacency -> 1-bit mask via wave32 ballot (adjacency read exactly once,
  // with a few rows of prefetch ahead of the streaming reads)
  for (int rr = 0; rr < 16; ++rr) {
    int row = slab + rr;
    if (rr + 4 < 16)
      __builtin_prefetch(adjf + (size_t)(slab + rr + 4) * NN + lane * 8, 0, 0);
#pragma unroll
    for (int wc = 0; wc < 8; ++wc) {
      float a = adjf[row * NN + wc * 32 + lane];
      unsigned long long b = __ballot(a > 0.5f);
      if (lane == 0) maskw[row * 8 + wc] = (unsigned int)b;
    }
  }
  __syncthreads();

  const float scale = 0.14142135623730951f; // 1/sqrt(50)
  v8f zero = {0.f, 0.f, 0.f, 0.f, 0.f, 0.f, 0.f, 0.f};

  for (int it = 0; it < 3; ++it) {
    // -------- q = tanh(V Wq^T + bq), k = tanh(V Wk^T + bk) (16x64 per wave) ----
    // B-tiles software-pipelined one ahead so loads land in the WMMA shadow.
    v8f qa[4], ka[4];
#pragma unroll
    for (int nt = 0; nt < 4; ++nt) { qa[nt] = zero; ka[nt] = zero; }
#pragma unroll
    for (int kb = 0; kb < VP; kb += 32) {
      v16h av  = ld_tile_a(valh, slab, VP, kb, lane);
      v16h bqn = ld_tile_b(wqh, 0, VP, kb, lane);
      v16h bkn = ld_tile_b(wkh, 0, VP, kb, lane);
#pragma unroll
      for (int nt = 0; nt < 4; ++nt) {
        v16h bqc = bqn, bkc = bkn;
        if (nt < 3) {
          bqn = ld_tile_b(wqh, (nt + 1) * 16, VP, kb, lane);
          bkn = ld_tile_b(wkh, (nt + 1) * 16, VP, kb, lane);
        }
        qa[nt] = wmma16(av, bqc, qa[nt]);
        ka[nt] = wmma16(av, bkc, ka[nt]);
      }
    }
#pragma unroll
    for (int nt = 0; nt < 4; ++nt) {
#pragma unroll
      for (int r = 0; r < 8; ++r) {
        int m = slab + r + 8 * hsel;
        int c = nt * 16 + nloc;
        qh[m * VP + c] = (_Float16)fast_tanh(qa[nt][r] + bqs[c]);
        kh[m * VP + c] = (_Float16)fast_tanh(ka[nt][r] + bks[c]);
      }
    }
    __syncthreads();

    // -------- S = q k^T  (full 16x256 slab in registers: 16 accum tiles) ------
    v8f S[16];
#pragma unroll
    for (int t = 0; t < 16; ++t) S[t] = zero;
#pragma unroll
    for (int kb = 0; kb < VP; kb += 32) {
      v16h aq = ld_tile_a(qh, slab, VP, kb, lane);
      v16h bn = ld_tile_b(kh, 0, VP, kb, lane);
#pragma unroll
      for (int t = 0; t < 16; ++t) {
        v16h bc = bn;
        if (t < 15) bn = ld_tile_b(kh, (t + 1) * 16, VP, kb, lane);
        S[t] = wmma16(aq, bc, S[t]);
      }
    }

    // -------- masked softmax over each row (in-register, shfl reductions) -----
#pragma unroll
    for (int r = 0; r < 8; ++r) {
      int m = slab + r + 8 * hsel;
      float mx = -3.0e38f;
#pragma unroll
      for (int t = 0; t < 16; ++t) {
        int c = t * 16 + nloc;
        unsigned int w = maskw[m * 8 + (c >> 5)];
        float e = S[t][r] * scale + (((w >> (c & 31)) & 1u) ? 0.f : -1000.f);
        S[t][r] = e;
        mx = fmaxf(mx, e);
      }
#pragma unroll
      for (int d = 8; d >= 1; d >>= 1) mx = fmaxf(mx, __shfl_xor(mx, d, 32));
      float sum = 0.f;
#pragma unroll
      for (int t = 0; t < 16; ++t) {
        float p = __expf(S[t][r] - mx);
        S[t][r] = p;
        sum += p;
      }
#pragma unroll
      for (int d = 8; d >= 1; d >>= 1) sum += __shfl_xor(sum, d, 32);
      float inv = 1.0f / sum;
#pragma unroll
      for (int t = 0; t < 16; ++t)
        alph[m * NN + t * 16 + nloc] = (_Float16)(S[t][r] * inv);
    }

    // -------- out = alpha @ V  (16x64 per wave, K=256) ------------------------
    v8f O[4];
#pragma unroll
    for (int nt = 0; nt < 4; ++nt) O[nt] = zero;
#pragma unroll
    for (int kb = 0; kb < NN; kb += 32) {
      v16h aa = ld_tile_a(alph, slab, NN, kb, lane);
      v16h bn = ld_tile_b(valt, 0, NN, kb, lane);
#pragma unroll
      for (int nt = 0; nt < 4; ++nt) {
        v16h bc = bn;
        if (nt < 3) bn = ld_tile_b(valt, (nt + 1) * 16, NN, kb, lane);
        O[nt] = wmma16(aa, bc, O[nt]);
      }
    }
    __syncthreads();  // everyone done reading valt/valh before overwrite

#pragma unroll
    for (int nt = 0; nt < 4; ++nt) {
#pragma unroll
      for (int r = 0; r < 8; ++r) {
        int m = slab + r + 8 * hsel;
        int c = nt * 16 + nloc;
        float x = O[nt][r];
        valh[m * VP + c] = (_Float16)x;
        valt[c * NN + m] = (_Float16)x;
        if (it == 2 && c < VD)
          out[(size_t)f * (NN * VD) + m * VD + c] = x;
      }
    }
    __syncthreads();  // new values visible before next iteration
  }
}

extern "C" void kernel_launch(void* const* d_in, const int* in_sizes, int n_in,
                              void* d_out, int out_size, void* d_ws, size_t ws_size,
                              hipStream_t stream) {
  const float* values = (const float*)d_in[0];
  const float* adjm   = (const float*)d_in[1];
  const float* Wq     = (const float*)d_in[2];
  const float* bq     = (const float*)d_in[3];
  const float* Wk     = (const float*)d_in[4];
  const float* bk     = (const float*)d_in[5];
  float* out = (float*)d_out;

  int F = in_sizes[0] / (NN * VD);   // 1024 graphs
  size_t smem_bytes = 287232;        // 280.5 KB of the 320 KB/WGP LDS
  gat_mp_kernel<<<dim3(F), dim3(512), smem_bytes, stream>>>(
      values, adjm, Wq, bq, Wk, bk, out);
}
